// PCASC_3332894622347
// MI455X (gfx1250) — compile-verified, matrix operand order checked by reading, not run on hardware
//
#include <hip/hip_runtime.h>
#include <hip/hip_bf16.h>
#include <cstddef>

// ---------------------------------------------------------------------------
// PVT-style cross attention block for MI455X (gfx1250, wave32, WMMA).
// All matmuls run through v_wmma_f32_16x16x32_f16 (f32 accumulate); memory
// tensors stay f32, operands converted to f16 in-register. Fragment loads are
// vectorized to b128 (global + LDS). Bandwidth-bound workload (~8.5 GFLOP,
// ~100-150 MB traffic -> ~6 us at 23.3 TB/s), so coalescing > peak math.
// ---------------------------------------------------------------------------

typedef __attribute__((ext_vector_type(16))) _Float16 v16h;
typedef __attribute__((ext_vector_type(8)))  _Float16 v8h;
typedef __attribute__((ext_vector_type(8)))  float    v8f;

#define DEV __device__ __forceinline__

DEV float4 ld4(const float* p) { return *(const float4*)p; }
DEV float4 add4(float4 a, float4 b) {
    return make_float4(a.x + b.x, a.y + b.y, a.z + b.z, a.w + b.w);
}
DEV v8h pack8(float4 a, float4 b) {
    v8h r;
    r[0] = (_Float16)a.x; r[1] = (_Float16)a.y; r[2] = (_Float16)a.z; r[3] = (_Float16)a.w;
    r[4] = (_Float16)b.x; r[5] = (_Float16)b.y; r[6] = (_Float16)b.z; r[7] = (_Float16)b.w;
    return r;
}
DEV v16h cat16(v8h lo, v8h hi) {
    v16h r;
#pragma unroll
    for (int j = 0; j < 8; ++j) { r[j] = lo[j]; r[j + 8] = hi[j]; }
    return r;
}

// Per-lane base K offset of the 16-bit fragment: runs [c0, c0+7] (elems 0-7)
// and [c0+16, c0+23] (elems 8-15), with c0 = 8 for the upper lane half.
DEV int frag_c0(int lane) { return (lane & 16) ? 8 : 0; }

// ---------------------------------------------------------------------------
// GEMM: Y[M,Nout] = (X (+ X2)) @ W[Nout,K]^T (+ bias).
// One wave computes a 16x64 strip: A fragment loaded once per k-chunk,
// 4 column tiles / 4 WMMAs. grid = (ceil(M/16), Nout/64).
// store_mode 0: row-major [M,Nout]; 1: [B, Nout, spatialN] (BCHW).
// ---------------------------------------------------------------------------
__global__ __launch_bounds__(32)
void pcasc_gemm(const float* __restrict__ X, const float* __restrict__ X2,
                const float* __restrict__ W, const float* __restrict__ bias,
                float* __restrict__ Y, int M, int K, int Nout,
                int store_mode, int spatialN)
{
    int mt = blockIdx.x, ng = blockIdx.y;
    int lane = threadIdx.x;
    int rowA = mt * 16 + (lane & 15);
    if (rowA > M - 1) rowA = M - 1;                 // clamp (stores guarded)
    int colbase = ng * 64;
    const float* xrow = X + (size_t)rowA * K;
    const bool hasX2 = (X2 != nullptr);             // uniform (SGPR) test

    v8f acc[4] = {}, z = {};
    acc[0] = z; acc[1] = z; acc[2] = z; acc[3] = z;

    for (int kk = 0; kk < K; kk += 32) {
        int c0 = kk + frag_c0(lane);
        float4 a0 = ld4(xrow + c0),      a1 = ld4(xrow + c0 + 4);
        float4 a2 = ld4(xrow + c0 + 16), a3 = ld4(xrow + c0 + 20);
        if (hasX2) {
            const float* x2row = X2 + (size_t)rowA * K;
            a0 = add4(a0, ld4(x2row + c0));
            a1 = add4(a1, ld4(x2row + c0 + 4));
            a2 = add4(a2, ld4(x2row + c0 + 16));
            a3 = add4(a3, ld4(x2row + c0 + 20));
        }
        v16h a = cat16(pack8(a0, a1), pack8(a2, a3));
#pragma unroll
        for (int t = 0; t < 4; ++t) {
            const float* wrow = W + (size_t)(colbase + t * 16 + (lane & 15)) * K;
            v16h b = cat16(pack8(ld4(wrow + c0),      ld4(wrow + c0 + 4)),
                           pack8(ld4(wrow + c0 + 16), ld4(wrow + c0 + 20)));
            acc[t] = __builtin_amdgcn_wmma_f32_16x16x32_f16(
                         false, a, false, b, (short)0, acc[t], false, false);
        }
    }

#pragma unroll
    for (int t = 0; t < 4; ++t) {
        int col = colbase + t * 16 + (lane & 15);
        float bv = bias ? bias[col] : 0.0f;
#pragma unroll
        for (int g = 0; g < 8; ++g) {
            int r = mt * 16 + g + ((lane & 16) ? 8 : 0);
            if (r < M) {
                float val = acc[t][g] + bv;
                if (store_mode == 0) {
                    Y[(size_t)r * Nout + col] = val;
                } else {
                    int bb = r / spatialN, n = r % spatialN;
                    Y[((size_t)bb * Nout + col) * spatialN + n] = val;
                }
            }
        }
    }
}

// ---------------------------------------------------------------------------
// Repack conv weight: srw[co][ci][khw] -> dst[co][khw][ci] (K-contiguous
// per output channel for the B fragment). 64*64*64 elements.
// ---------------------------------------------------------------------------
__global__ __launch_bounds__(256)
void pcasc_repack_srw(const float* __restrict__ src, float* __restrict__ dst)
{
    int idx = blockIdx.x * 256 + threadIdx.x;       // writes coalesced
    int ci  = idx & 63;
    int khw = (idx >> 6) & 63;
    int co  = idx >> 12;
    dst[idx] = src[((size_t)co * 64 + ci) * 64 + khw];
}

// ---------------------------------------------------------------------------
// SR conv (8x8 stride 8) as GEMM: M=B*196 patches, K=4096 (khw-major,
// ci-minor), Nout=64. One wave -> 16x64 strip, 4 WMMAs per k-chunk.
// ---------------------------------------------------------------------------
__global__ __launch_bounds__(32)
void pcasc_srconv(const float* __restrict__ Xkv,   // [B, 12544, 64]
                  const float* __restrict__ Wt,    // repacked [co][khw][ci]
                  const float* __restrict__ bias,  // [64]
                  float* __restrict__ Y,           // [B*196, 64]
                  int B)
{
    const int C = 64, Wimg = 112, OW = 14;
    int mt = blockIdx.x;
    int lane = threadIdx.x;
    int M = B * 196;
    int row = mt * 16 + (lane & 15);
    if (row > M - 1) row = M - 1;
    int bb = row / 196, p = row % 196;
    int oh = p / OW, ow = p % OW;

    v8f acc[4] = {};
    for (int kk = 0; kk < 4096; kk += 32) {
        int khw = kk >> 6;                           // constant per chunk
        int kh = khw >> 3, kw = khw & 7;
        int nsp = (oh * 8 + kh) * Wimg + (ow * 8 + kw);
        const float* xrow = Xkv + ((size_t)bb * 12544 + nsp) * C;
        int c0 = (kk & 63) + frag_c0(lane);          // ci base
        v16h a = cat16(pack8(ld4(xrow + c0),      ld4(xrow + c0 + 4)),
                       pack8(ld4(xrow + c0 + 16), ld4(xrow + c0 + 20)));
#pragma unroll
        for (int t = 0; t < 4; ++t) {
            const float* wrow = Wt + ((size_t)(t * 16 + (lane & 15)) * 64 + khw) * 64;
            v16h b = cat16(pack8(ld4(wrow + c0),      ld4(wrow + c0 + 4)),
                           pack8(ld4(wrow + c0 + 16), ld4(wrow + c0 + 20)));
            acc[t] = __builtin_amdgcn_wmma_f32_16x16x32_f16(
                         false, a, false, b, (short)0, acc[t], false, false);
        }
    }

#pragma unroll
    for (int t = 0; t < 4; ++t) {
        int co = t * 16 + (lane & 15);
        float bv = bias[co];
#pragma unroll
        for (int g = 0; g < 8; ++g) {
            int r = mt * 16 + g + ((lane & 16) ? 8 : 0);
            if (r < M) Y[(size_t)r * 64 + co] = acc[t][g] + bv;
        }
    }
}

// ---------------------------------------------------------------------------
// LayerNorm over C=64, one thread per row (only 392 rows).
// ---------------------------------------------------------------------------
__global__ void pcasc_ln(const float* __restrict__ X, const float* __restrict__ g,
                         const float* __restrict__ b, float* __restrict__ Y, int rows)
{
    int r = blockIdx.x * blockDim.x + threadIdx.x;
    if (r >= rows) return;
    const float* x = X + (size_t)r * 64;
    float m = 0.f;
    for (int i = 0; i < 64; ++i) m += x[i];
    m *= (1.f / 64.f);
    float v = 0.f;
    for (int i = 0; i < 64; ++i) { float d = x[i] - m; v += d * d; }
    v *= (1.f / 64.f);
    float inv = rsqrtf(v + 1e-5f);
    float* y = Y + (size_t)r * 64;
    for (int i = 0; i < 64; ++i) y[i] = (x[i] - m) * inv * g[i] + b[i];
}

// ---------------------------------------------------------------------------
// Fused attention: scores (WMMA) -> softmax (f32, LDS) -> PV (WMMA).
// Block = 64 threads (2 waves); each wave owns 16 query rows of one (b,h).
// grid = (B*HEADS, N/32).
// ---------------------------------------------------------------------------
#define NKV      196
#define NKV_P16  208   // 13 * 16
#define NKV_P32  224   // 7  * 32

__global__ __launch_bounds__(64)
void pcasc_attn(const float* __restrict__ Q,   // [B*12544, 64] head-packed
                const float* __restrict__ KV,  // [B*196, 128] (k | v)
                float* __restrict__ O,         // [B*12544, 64]
                int B)
{
    __shared__ _Float16 Klds[NKV_P32][8];      // [kv][d] : row = 16B
    __shared__ _Float16 Vlds[8][NKV_P32];      // [d][kv] : K-contiguous
    __shared__ float    S[2][16][NKV_P16];
    __shared__ _Float16 P[2][16][NKV_P32];     // row = 448B (16B aligned)

    int bh = blockIdx.x;
    int bb = bh >> 3, h = bh & 7;
    int qt = blockIdx.y;
    int tid = threadIdx.x;
    int wave = tid >> 5, lane = tid & 31;
    const v8h zero8 = {};

    // Stage K/V head slices into LDS as f16 (zero padded).
    for (int i = tid; i < NKV_P32; i += 64) {
        const float* src = KV + ((size_t)bb * 196 + (i < NKV ? i : 0)) * 128;
#pragma unroll
        for (int d = 0; d < 8; ++d) {
            float kf = (i < NKV) ? src[h * 8 + d]      : 0.f;
            float vf = (i < NKV) ? src[64 + h * 8 + d] : 0.f;
            Klds[i][d] = (_Float16)kf;
            Vlds[d][i] = (_Float16)vf;
        }
    }
    __syncthreads();

    // Q fragment, scale folded in (1/sqrt(8)). K=8 lives in positions 0..7,
    // so the upper lane half (K 8..15 / 24..31) is structurally zero.
    const float scale = 0.35355339059327373f;
    int qrow = qt * 32 + wave * 16 + (lane & 15);
    v16h a;
    if (!(lane & 16)) {
        const float* qsrc = Q + ((size_t)bb * 12544 + qrow) * 64 + h * 8;
        float4 q0 = ld4(qsrc), q1 = ld4(qsrc + 4);
        q0.x *= scale; q0.y *= scale; q0.z *= scale; q0.w *= scale;
        q1.x *= scale; q1.y *= scale; q1.z *= scale; q1.w *= scale;
        a = cat16(pack8(q0, q1), zero8);
    } else {
        a = cat16(zero8, zero8);
    }

    // Scores: 13 column tiles of 16 kv positions.
    for (int t = 0; t < 13; ++t) {
        int kvcol = t * 16 + (lane & 15);
        v16h bf;
        if (!(lane & 16)) {
            v8h krow = *(const v8h*)&Klds[kvcol][0];
            bf = cat16(krow, zero8);
        } else {
            bf = cat16(zero8, zero8);
        }
        v8f sc = {};
        sc = __builtin_amdgcn_wmma_f32_16x16x32_f16(
                 false, a, false, bf, (short)0, sc, false, false);
#pragma unroll
        for (int g = 0; g < 8; ++g) {
            int r = g + ((lane & 16) ? 8 : 0);
            S[wave][r][t * 16 + (lane & 15)] = sc[g];
        }
    }
    __syncthreads();

    // Softmax (f32): lanes 0-15 each own one query row, true 196 columns.
    if (lane < 16) {
        float mx = -1e30f;
        for (int i = 0; i < NKV; ++i) mx = fmaxf(mx, S[wave][lane][i]);
        float sum = 0.f;
        for (int i = 0; i < NKV; ++i) {
            float e = __expf(S[wave][lane][i] - mx);
            S[wave][lane][i] = e;
            sum += e;
        }
        float inv = 1.f / sum;
        for (int i = 0; i < NKV_P32; ++i)
            P[wave][lane][i] = (i < NKV) ? (_Float16)(S[wave][lane][i] * inv)
                                         : (_Float16)0.f;
    }
    __syncthreads();

    // O = P @ V : K = 196 padded to 224 (7 chunks of 32); V cols 8-15 zero.
    v8f oacc = {};
    int prow = lane & 15;
    int ncol = lane & 15;
    for (int t = 0; t < 7; ++t) {
        int k0 = t * 32 + frag_c0(lane);
        v16h pa = cat16(*(const v8h*)&P[wave][prow][k0],
                        *(const v8h*)&P[wave][prow][k0 + 16]);
        v16h vb;
        if (ncol < 8) {
            vb = cat16(*(const v8h*)&Vlds[ncol][k0],
                       *(const v8h*)&Vlds[ncol][k0 + 16]);
        } else {
            vb = cat16(zero8, zero8);
        }
        oacc = __builtin_amdgcn_wmma_f32_16x16x32_f16(
                   false, pa, false, vb, (short)0, oacc, false, false);
    }
    if (ncol < 8) {
#pragma unroll
        for (int g = 0; g < 8; ++g) {
            int r = qt * 32 + wave * 16 + g + ((lane & 16) ? 8 : 0);
            O[((size_t)bb * 12544 + r) * 64 + h * 8 + ncol] = oacc[g];
        }
    }
}

// ---------------------------------------------------------------------------
// [B,C,H*W] -> [B*H*W, C] via LDS-tiled transpose; both phases coalesced.
// grid = (12544/64, B), block = 256.
// ---------------------------------------------------------------------------
__global__ __launch_bounds__(256)
void pcasc_nchw_to_nlc(const float* __restrict__ X, float* __restrict__ Y)
{
    __shared__ float tile[64][65];
    int n0 = blockIdx.x * 64;
    int b  = blockIdx.y;
    for (int i = threadIdx.x; i < 4096; i += 256) {
        int c = i >> 6, n = i & 63;
        tile[c][n] = X[((size_t)b * 64 + c) * 12544 + n0 + n];
    }
    __syncthreads();
    for (int i = threadIdx.x; i < 4096; i += 256) {
        int n = i >> 6, c = i & 63;
        Y[((size_t)b * 12544 + n0 + n) * 64 + c] = tile[c][n];
    }
}

// ---------------------------------------------------------------------------
// Host orchestration.
// d_in order: x1, x2, then per module {qw,kvw,pw,pb,srw,srb,lng,lnb} for
// sa1(2), ca12(10), sa2(18), ca21(26), then pw(34), pb(35).
// ---------------------------------------------------------------------------
extern "C" void kernel_launch(void* const* d_in, const int* in_sizes, int n_in,
                              void* d_out, int out_size, void* d_ws, size_t ws_size,
                              hipStream_t stream)
{
    (void)in_sizes; (void)n_in; (void)out_size; (void)ws_size;
    const int B = 2, N = 12544, NB = B * N, NKVR = B * 196;
    const float* x1 = (const float*)d_in[0];
    const float* x2 = (const float*)d_in[1];
    auto W = [&](int i) { return (const float*)d_in[i]; };

    float* ws = (float*)d_ws;
    size_t off = 0;
    auto alloc = [&](size_t nfl) { float* p = ws + off; off += nfl; return p; };
    float* xf1   = alloc((size_t)NB * 64);
    float* xf2   = alloc((size_t)NB * 64);
    float* bufA  = alloc((size_t)NB * 64);
    float* bufB  = alloc((size_t)NB * 64);
    float* bufC  = alloc((size_t)NB * 64);
    float* bufD  = alloc((size_t)NB * 64);
    float* tmpQ  = alloc((size_t)NB * 64);
    float* attnO = alloc((size_t)NB * 64);
    float* kvsrc = alloc((size_t)NKVR * 64);
    float* kvln  = alloc((size_t)NKVR * 64);
    float* kvprj = alloc((size_t)NKVR * 128);
    float* srw_t = alloc((size_t)64 * 64 * 64);

    pcasc_nchw_to_nlc<<<dim3(N / 64, B), 256, 0, stream>>>(x1, xf1);
    pcasc_nchw_to_nlc<<<dim3(N / 64, B), 256, 0, stream>>>(x2, xf2);

    auto run_attn = [&](const float* xq, const float* xkv, int base, float* out) {
        const float* qw  = W(base + 0);
        const float* kvw = W(base + 1);
        const float* pw  = W(base + 2);
        const float* pb  = W(base + 3);
        const float* srw = W(base + 4);
        const float* srb = W(base + 5);
        const float* lng = W(base + 6);
        const float* lnb = W(base + 7);
        // q = xq @ qw^T
        pcasc_gemm<<<dim3(NB / 16, 1), 32, 0, stream>>>(
            xq, nullptr, qw, nullptr, tmpQ, NB, 64, 64, 0, 0);
        // spatial reduction conv + bias (weights repacked K-contiguous)
        pcasc_repack_srw<<<(64 * 64 * 64) / 256, 256, 0, stream>>>(srw, srw_t);
        pcasc_srconv<<<dim3(25, 1), 32, 0, stream>>>(xkv, srw_t, srb, kvsrc, B);
        // layernorm
        pcasc_ln<<<(NKVR + 63) / 64, 64, 0, stream>>>(kvsrc, lng, lnb, kvln, NKVR);
        // kv = ln @ kvw^T  ([B*196, 128])
        pcasc_gemm<<<dim3(25, 2), 32, 0, stream>>>(
            kvln, nullptr, kvw, nullptr, kvprj, NKVR, 64, 128, 0, 0);
        // fused softmax attention
        pcasc_attn<<<dim3(B * 8, N / 32), 64, 0, stream>>>(tmpQ, kvprj, attnO, B);
        // out = (o + xq) @ pw^T + pb
        pcasc_gemm<<<dim3(NB / 16, 1), 32, 0, stream>>>(
            attnO, xq, pw, pb, out, NB, 64, 64, 0, 0);
    };

    run_attn(xf1, xf1,  2, bufA);   // sa1: self attention on x1
    run_attn(xf2, bufA, 10, bufB);  // ca12: q from x2, kv from a
    run_attn(bufB, bufB, 18, bufC); // sa2: self attention on b
    run_attn(bufA, bufC, 26, bufD); // ca21: q from a, kv from c

    // fuse = d @ pw^T + pb, stored transposed to [B,C,H,W]
    pcasc_gemm<<<dim3(NB / 16, 1), 32, 0, stream>>>(
        bufD, nullptr, W(34), W(35), (float*)d_out, NB, 64, 64, 1, N);
}
